// GNN_56418690401074
// MI455X (gfx1250) — matrix operand order repaired and checked by reading, not compiled, not run on hardware
//
#include <hip/hip_runtime.h>
#include <math.h>

// ---------------------------------------------------------------------------
// MI455X (gfx1250) implementation of the hierarchical GAT reference.
//  - dense GEMMs via v_wmma_f32_16x16x32_bf16 (bf16 operands, f32 accum)
//    * all GEMM M dims are multiples of 64 by construction -> no row guards
//    * float4 (b128) global staging, packed b64 LDS stores, BK=64 stages
//  - edge softmax via ordered-uint atomicMax + f32 atomicAdd (L2 atomics)
//  - topk set-selection per graph (order-invariant downstream)
// Workspace usage ~142 MB.
// ---------------------------------------------------------------------------

static const int GB    = 64;      // graphs in batch
static const int NPER0 = 512;     // nodes per graph, level 0
static const int NE    = 131072;  // edges (constant across levels; mask shrinks)
static const int HEADS = 3;
static const int CDIM  = 128;
static const int GWID  = HEADS * CDIM;  // 384
static const int NMAX  = GB * NPER0;    // 32768

typedef __attribute__((ext_vector_type(16))) __bf16 v16bf;
typedef __attribute__((ext_vector_type(8)))  float  v8f;

// ---------------- helpers ----------------
__device__ __forceinline__ unsigned f2bf_u(float x) {   // -> bf16 bits in [15:0]
  unsigned u = __float_as_uint(x);
  unsigned r = u + 0x7fffu + ((u >> 16) & 1u);          // round-to-nearest-even
  return r >> 16;
}
__device__ __forceinline__ unsigned ford_enc(float x) { // monotone float->uint
  unsigned u = __float_as_uint(x);
  return (u & 0x80000000u) ? ~u : (u | 0x80000000u);
}
__device__ __forceinline__ float ford_dec(unsigned e) {
  unsigned u = (e & 0x80000000u) ? (e ^ 0x80000000u) : ~e;
  return __uint_as_float(u);
}

// ---------------- WMMA GEMM: C = A[MxK] * B[KxN] (+bias, opt relu) ----------
// Requires: M % 64 == 0, N % 64 == 0, K % 64 == 0 (true for every call here).
#define GEMM_BM 64
#define GEMM_BN 64
#define GEMM_BK 64
#define LDP (GEMM_BK + 4)   // 68 ushorts: 136B rows -> 8B-aligned quad stores

__device__ __forceinline__ v16bf load_fragA(const unsigned short* lA, int row0,
                                            int koff, int lane) {
  union { v16bf v; unsigned u[8]; } f;
  const unsigned short* rp = lA + (row0 + (lane & 15)) * LDP + koff;
  int kb = (lane >> 4) * 8;
#pragma unroll
  for (int v = 0; v < 4; ++v) f.u[v]     = *(const unsigned*)(rp + kb + 2 * v);
#pragma unroll
  for (int v = 0; v < 4; ++v) f.u[4 + v] = *(const unsigned*)(rp + kb + 16 + 2 * v);
  return f.v;
}
__device__ __forceinline__ v16bf load_fragB(const unsigned short* lBt, int col0,
                                            int koff, int lane) {
  union { v16bf v; unsigned u[8]; } f;
  const unsigned short* cp = lBt + (col0 + (lane & 15)) * LDP + koff;
  int kb = (lane >> 4) * 8;
#pragma unroll
  for (int v = 0; v < 4; ++v) f.u[v]     = *(const unsigned*)(cp + kb + 2 * v);
#pragma unroll
  for (int v = 0; v < 4; ++v) f.u[4 + v] = *(const unsigned*)(cp + kb + 16 + 2 * v);
  return f.v;
}

__global__ __launch_bounds__(128) void wmma_gemm_bf16(
    const float* __restrict__ A, const float* __restrict__ B,
    const float* __restrict__ bias, float* __restrict__ C,
    int M, int N, int K, int do_relu)
{
  __shared__ unsigned short lA[GEMM_BM * LDP];
  __shared__ unsigned short lB[GEMM_BN * LDP];   // B stored transposed: [n][k]
  const int tid  = threadIdx.x;
  const int lane = tid & 31;
  const int wave = tid >> 5;
  const int wr = wave >> 1, wc = wave & 1;       // 2x2 wave grid
  const int brow = blockIdx.y * GEMM_BM;
  const int bcol = blockIdx.x * GEMM_BN;

  union { v8f v; float f[8]; } acc[2][2];
#pragma unroll
  for (int i = 0; i < 2; ++i)
#pragma unroll
    for (int j = 0; j < 2; ++j)
#pragma unroll
      for (int e = 0; e < 8; ++e) acc[i][j].f[e] = 0.f;

  for (int k0 = 0; k0 < K; k0 += GEMM_BK) {
    // A tile 64x64 fp32: b128 loads, packed b64 LDS stores (no guards: M%64==0)
#pragma unroll
    for (int it = 0; it < 8; ++it) {
      int idx = tid + it * 128;                  // float4 index, 1024 total
      int r = idx >> 4, c4 = (idx & 15) * 4;
      const float4 v = *(const float4*)(A + (size_t)(brow + r) * K + k0 + c4);
      uint2 p;
      p.x = f2bf_u(v.x) | (f2bf_u(v.y) << 16);
      p.y = f2bf_u(v.z) | (f2bf_u(v.w) << 16);
      *(uint2*)(lA + r * LDP + c4) = p;
    }
    // B tile 64(k)x64(n) fp32: b128 loads along n, transposed b16 LDS stores
#pragma unroll
    for (int it = 0; it < 8; ++it) {
      int idx = tid + it * 128;
      int kk = idx >> 4, n4 = (idx & 15) * 4;
      const float4 v = *(const float4*)(B + (size_t)(k0 + kk) * N + bcol + n4);
      lB[(n4 + 0) * LDP + kk] = (unsigned short)f2bf_u(v.x);
      lB[(n4 + 1) * LDP + kk] = (unsigned short)f2bf_u(v.y);
      lB[(n4 + 2) * LDP + kk] = (unsigned short)f2bf_u(v.z);
      lB[(n4 + 3) * LDP + kk] = (unsigned short)f2bf_u(v.w);
    }
    __syncthreads();
#pragma unroll
    for (int kh = 0; kh < 2; ++kh) {             // two 16x16x32 sub-steps
      v16bf af0 = load_fragA(lA, wr * 32,      kh * 32, lane);
      v16bf af1 = load_fragA(lA, wr * 32 + 16, kh * 32, lane);
      v16bf bf0 = load_fragB(lB, wc * 32,      kh * 32, lane);
      v16bf bf1 = load_fragB(lB, wc * 32 + 16, kh * 32, lane);
      acc[0][0].v = __builtin_amdgcn_wmma_f32_16x16x32_bf16(false, af0, false, bf0, (short)0, acc[0][0].v, false, false);
      acc[0][1].v = __builtin_amdgcn_wmma_f32_16x16x32_bf16(false, af0, false, bf1, (short)0, acc[0][1].v, false, false);
      acc[1][0].v = __builtin_amdgcn_wmma_f32_16x16x32_bf16(false, af1, false, bf0, (short)0, acc[1][0].v, false, false);
      acc[1][1].v = __builtin_amdgcn_wmma_f32_16x16x32_bf16(false, af1, false, bf1, (short)0, acc[1][1].v, false, false);
    }
    __syncthreads();
  }
  const int rb = (lane >> 4) * 8;                // f32 C/D lane layout
  const int nc = lane & 15;
#pragma unroll
  for (int i = 0; i < 2; ++i)
#pragma unroll
    for (int j = 0; j < 2; ++j) {
      int col = bcol + wc * 32 + j * 16 + nc;
      float bv = bias ? bias[col] : 0.f;
#pragma unroll
      for (int e = 0; e < 8; ++e) {
        int row = brow + wr * 32 + i * 16 + rb + e;
        float v = acc[i][j].f[e] + bv;
        if (do_relu) v = fmaxf(v, 0.f);
        C[(size_t)row * N + col] = v;
      }
    }
}

// ---------------- small utility kernels ----------------
__global__ void fill_f32(float* p, float v, int n) {
  int i = blockIdx.x * blockDim.x + threadIdx.x;
  if (i < n) p[i] = v;
}
__global__ void bias_fill(float* out, const float* __restrict__ b, int n, int w) {
  int i = blockIdx.x * blockDim.x + threadIdx.x;
  if (i < n) out[i] = b[i % w];
}
__global__ void attn_init(unsigned* mx, float* den, int n3) {
  int i = blockIdx.x * blockDim.x + threadIdx.x;
  if (i < n3) { mx[i] = ford_enc(-3.0e38f); den[i] = 0.f; }
}
__global__ void edge_init(const int* __restrict__ ei, int* s, int* d, int* m, int E) {
  int e = blockIdx.x * blockDim.x + threadIdx.x;
  if (e < E) { s[e] = ei[e]; d[e] = ei[E + e]; m[e] = 1; }
}

// e_src/e_dst per (node, head): sum_c h[n,h,c]*a[h,c]
__global__ void attn_coeffs(const float* __restrict__ h, const float* __restrict__ a_src,
                            const float* __restrict__ a_dst, float* es, float* ed, int n) {
  int i = blockIdx.x * blockDim.x + threadIdx.x;
  if (i >= n * HEADS) return;
  int node = i / HEADS, hd = i % HEADS;
  const float* hp = h + (size_t)node * GWID + hd * CDIM;
  const float* as = a_src + hd * CDIM;
  const float* ad = a_dst + hd * CDIM;
  float s = 0.f, d = 0.f;
  for (int c = 0; c < CDIM; ++c) { float v = hp[c]; s += v * as[c]; d += v * ad[c]; }
  es[i] = s; ed[i] = d;
}

__global__ void attn_max(const int* __restrict__ src, const int* __restrict__ dst,
                         const int* __restrict__ msk, const float* __restrict__ es,
                         const float* __restrict__ ed, unsigned* mx, int E, int n) {
  int e = blockIdx.x * blockDim.x + threadIdx.x;
  if (e >= E + n) return;
  int s, d;
  if (e < E) { if (!msk[e]) return; s = src[e]; d = dst[e]; } else { s = d = e - E; }
#pragma unroll
  for (int hd = 0; hd < HEADS; ++hd) {
    float l = es[s * HEADS + hd] + ed[d * HEADS + hd];
    l = (l > 0.f) ? l : 0.2f * l;                          // leaky_relu(0.2)
    atomicMax(&mx[d * HEADS + hd], ford_enc(l));
  }
}

__global__ void attn_expsum(const int* __restrict__ src, const int* __restrict__ dst,
                            const int* __restrict__ msk, const float* __restrict__ es,
                            const float* __restrict__ ed, const unsigned* __restrict__ mx,
                            float* ex, float* den, int E, int n) {
  int e = blockIdx.x * blockDim.x + threadIdx.x;
  if (e >= E + n) return;
  int s, d;
  if (e < E) { if (!msk[e]) return; s = src[e]; d = dst[e]; } else { s = d = e - E; }
#pragma unroll
  for (int hd = 0; hd < HEADS; ++hd) {
    float l = es[s * HEADS + hd] + ed[d * HEADS + hd];
    l = (l > 0.f) ? l : 0.2f * l;
    float v = expf(l - ford_dec(mx[d * HEADS + hd]));
    ex[e * HEADS + hd] = v;
    atomicAdd(&den[d * HEADS + hd], v);
  }
}

// one thread per (edge, head, 4-channel chunk): float4 gather + 4 atomics
__global__ void attn_aggregate(const int* __restrict__ src, const int* __restrict__ dst,
                               const int* __restrict__ msk, const float* __restrict__ ex,
                               const float* __restrict__ den, const float* __restrict__ h,
                               float* out, int E, int n) {
  int i = blockIdx.x * blockDim.x + threadIdx.x;
  int tot = (E + n) * HEADS * (CDIM / 4);
  if (i >= tot) return;
  int chunk = i & (CDIM / 4 - 1);
  int eh = i / (CDIM / 4);
  int e = eh / HEADS, hd = eh % HEADS;
  int s, d;
  if (e < E) { if (!msk[e]) return; s = src[e]; d = dst[e]; } else { s = d = e - E; }
  float alpha = ex[e * HEADS + hd] / (den[d * HEADS + hd] + 1e-16f);
  int c0 = chunk * 4;
  const float4 hv = *(const float4*)(h + (size_t)s * GWID + hd * CDIM + c0);
  float* op = out + (size_t)d * GWID + hd * CDIM + c0;
  atomicAdd(&op[0], alpha * hv.x);
  atomicAdd(&op[1], alpha * hv.y);
  atomicAdd(&op[2], alpha * hv.z);
  atomicAdd(&op[3], alpha * hv.w);
}

// score = tanh(x.w / ||w||)
__global__ void pool_score(const float* __restrict__ x, const float* __restrict__ w,
                           float* score, int n) {
  int i = blockIdx.x * blockDim.x + threadIdx.x;
  if (i >= n) return;
  float ss = 0.f, dot = 0.f;
  const float* xp = x + (size_t)i * CDIM;
  for (int c = 0; c < CDIM; ++c) { ss += w[c] * w[c]; dot += xp[c] * w[c]; }
  score[i] = tanhf(dot * rsqrtf(ss));
}

// per-graph top-k set selection + compaction (order-invariant downstream)
__global__ __launch_bounds__(512) void topk_select(const float* __restrict__ score,
                                                   int per, int k, int* newid, int* perm) {
  __shared__ float ls[512];
  __shared__ int   pfx[512];
  int g = blockIdx.x, t = threadIdx.x;
  int base = g * per;
  if (t < per) ls[t] = score[base + t];
  __syncthreads();
  int kept = 0;
  if (t < per) {
    float my = ls[t];
    int rank = 0;
    for (int j = 0; j < per; ++j) {
      float o = ls[j];
      rank += (o > my) || (o == my && j < t);
    }
    kept = (rank < k) ? 1 : 0;
  }
  pfx[t] = kept;
  __syncthreads();
  for (int off = 1; off < 512; off <<= 1) {      // inclusive Hillis-Steele scan
    int add = (t >= off) ? pfx[t - off] : 0;
    __syncthreads();
    pfx[t] += add;
    __syncthreads();
  }
  if (t < per) {
    int nid = kept ? (g * k + pfx[t] - 1) : -1;
    newid[base + t] = nid;
    if (kept) perm[nid] = base + t;
  }
}

__global__ void gather_gate(const float* __restrict__ x, const float* __restrict__ score,
                            const int* __restrict__ perm, float* xn, int nnew) {
  int i = blockIdx.x * blockDim.x + threadIdx.x;
  if (i >= nnew * CDIM) return;
  int r = i / CDIM, c = i % CDIM;
  int old = perm[r];
  xn[i] = x[(size_t)old * CDIM + c] * score[old];
}

__global__ void remap_edges(const int* si, const int* di, const int* mi,
                            const int* __restrict__ newid,
                            int* so, int* dO, int* mo, int E) {
  int e = blockIdx.x * blockDim.x + threadIdx.x;
  if (e >= E) return;
  int v = mi[e];
  int ns = newid[si[e]], nd = newid[di[e]];
  int valid = (v && ns >= 0 && nd >= 0) ? 1 : 0;
  so[e] = valid ? ns : 0;
  dO[e] = valid ? nd : 0;
  mo[e] = valid;
}

// cat([max, mean]) per graph, accumulated into z (z += ...)
__global__ __launch_bounds__(128) void gpool_acc(const float* __restrict__ x, float* z, int k) {
  int g = blockIdx.x, c = threadIdx.x;
  float mx = -3.0e38f, sm = 0.f;
  const float* xp = x + (size_t)g * k * CDIM + c;
  for (int i = 0; i < k; ++i) { float v = xp[(size_t)i * CDIM]; mx = fmaxf(mx, v); sm += v; }
  z[g * 2 * CDIM + c]        += mx;
  z[g * 2 * CDIM + CDIM + c] += sm / (float)k;
}

// ---------------------------------------------------------------------------
static inline int cdiv(int a, int b) { return (a + b - 1) / b; }

extern "C" void kernel_launch(void* const* d_in, const int* in_sizes, int n_in,
                              void* d_out, int out_size, void* d_ws, size_t ws_size,
                              hipStream_t stream) {
  const float* x0 = (const float*)d_in[0];
  const int* edge_index = (const int*)d_in[2];
  const float* Wg[3]   = {(const float*)d_in[4],  (const float*)d_in[11], (const float*)d_in[18]};
  const float* asr[3]  = {(const float*)d_in[5],  (const float*)d_in[12], (const float*)d_in[19]};
  const float* adt[3]  = {(const float*)d_in[6],  (const float*)d_in[13], (const float*)d_in[20]};
  const float* bgt[3]  = {(const float*)d_in[7],  (const float*)d_in[14], (const float*)d_in[21]};
  const float* Whd[3]  = {(const float*)d_in[8],  (const float*)d_in[15], (const float*)d_in[22]};
  const float* bhd[3]  = {(const float*)d_in[9],  (const float*)d_in[16], (const float*)d_in[23]};
  const float* pwv[3]  = {(const float*)d_in[10], (const float*)d_in[17], (const float*)d_in[24]};
  const float* Wl1 = (const float*)d_in[25];
  const float* bl1 = (const float*)d_in[26];
  const float* Wl2 = (const float*)d_in[27];
  const float* bl2 = (const float*)d_in[28];

  // ---- workspace layout (256B aligned slices) ----
  size_t off = 0;
  auto alloc = [&](size_t nbytes) -> char* {
    char* r = (char*)d_ws + off;
    off += (nbytes + 255) & ~(size_t)255;
    return r;
  };
  float*    hbuf  = (float*)alloc((size_t)NMAX * GWID * 4);   // GAT hidden h
  float*    gout  = (float*)alloc((size_t)NMAX * GWID * 4);   // GAT output (bias-seeded)
  float*    feat0 = (float*)alloc((size_t)NMAX * CDIM * 4);   // post-Wh features
  float*    feat1 = (float*)alloc((size_t)NMAX * CDIM * 4);   // pooled features
  float*    esrc  = (float*)alloc((size_t)NMAX * HEADS * 4);
  float*    edst  = (float*)alloc((size_t)NMAX * HEADS * 4);
  unsigned* mxb   = (unsigned*)alloc((size_t)NMAX * HEADS * 4);
  float*    denb  = (float*)alloc((size_t)NMAX * HEADS * 4);
  float*    exb   = (float*)alloc((size_t)(NE + NMAX) * HEADS * 4);
  float*    scr   = (float*)alloc((size_t)NMAX * 4);
  int*      newid = (int*)alloc((size_t)NMAX * 4);
  int*      perm  = (int*)alloc((size_t)NMAX * 4);
  int*      sA = (int*)alloc((size_t)NE * 4);
  int*      dA = (int*)alloc((size_t)NE * 4);
  int*      mA = (int*)alloc((size_t)NE * 4);
  int*      sB = (int*)alloc((size_t)NE * 4);
  int*      dB = (int*)alloc((size_t)NE * 4);
  int*      mB = (int*)alloc((size_t)NE * 4);
  float*    zacc = (float*)alloc((size_t)GB * 2 * CDIM * 4);
  float*    tmlp = (float*)alloc((size_t)GB * 512 * 4);

  const int nlev[4] = {GB * 512, GB * 410, GB * 205, GB * 41};
  const int perl[3] = {512, 410, 205};
  const int keep[3] = {410, 205, 41};

  const int T = 256;
  fill_f32<<<cdiv(GB * 2 * CDIM, T), T, 0, stream>>>(zacc, 0.f, GB * 2 * CDIM);
  edge_init<<<cdiv(NE, T), T, 0, stream>>>(edge_index, sA, dA, mA, NE);

  int *sc = sA, *dc = dA, *mc = mA, *sn = sB, *dn = dB, *mn = mB;
  const float* featIn = x0;

  for (int l = 0; l < 3; ++l) {
    const int n = nlev[l];
    // 1) h = x @ W  (no bias)  -> hbuf [n, 384]     (WMMA bf16)
    wmma_gemm_bf16<<<dim3(GWID / 64, n / 64), 128, 0, stream>>>(
        featIn, Wg[l], nullptr, hbuf, n, GWID, CDIM, 0);
    // 2) attention coefficients
    attn_coeffs<<<cdiv(n * HEADS, T), T, 0, stream>>>(hbuf, asr[l], adt[l], esrc, edst, n);
    // 3) init max/den, seed gout with bias
    attn_init<<<cdiv(n * HEADS, T), T, 0, stream>>>(mxb, denb, n * HEADS);
    bias_fill<<<cdiv(n * GWID, T), T, 0, stream>>>(gout, bgt[l], n * GWID, GWID);
    // 4) edge softmax (self-loops = n virtual edges)
    attn_max<<<cdiv(NE + n, T), T, 0, stream>>>(sc, dc, mc, esrc, edst, mxb, NE, n);
    attn_expsum<<<cdiv(NE + n, T), T, 0, stream>>>(sc, dc, mc, esrc, edst, mxb, exb, denb, NE, n);
    attn_aggregate<<<cdiv((NE + n) * HEADS * (CDIM / 4), T), T, 0, stream>>>(
        sc, dc, mc, exb, denb, hbuf, gout, NE, n);
    // 5) h2 = relu(gout @ Wh + bh) -> feat0 [n, 128]   (WMMA bf16)
    wmma_gemm_bf16<<<dim3(CDIM / 64, n / 64), 128, 0, stream>>>(
        gout, Whd[l], bhd[l], feat0, n, CDIM, GWID, 1);
    // 6) topk pool
    pool_score<<<cdiv(n, T), T, 0, stream>>>(feat0, pwv[l], scr, n);
    topk_select<<<GB, 512, 0, stream>>>(scr, perl[l], keep[l], newid, perm);
    const int nnew = nlev[l + 1];
    gather_gate<<<cdiv(nnew * CDIM, T), T, 0, stream>>>(feat0, scr, perm, feat1, nnew);
    remap_edges<<<cdiv(NE, T), T, 0, stream>>>(sc, dc, mc, newid, sn, dn, mn, NE);
    // 7) global pool: z += cat(max, mean)
    gpool_acc<<<GB, CDIM, 0, stream>>>(feat1, zacc, keep[l]);
    // swap edge ping-pong; next layer input = pooled features
    int* t1 = sc; sc = sn; sn = t1;
    int* t2 = dc; dc = dn; dn = t2;
    int* t3 = mc; mc = mn; mn = t3;
    featIn = feat1;
  }

  // final MLP: relu(z @ Wl1 + bl1) @ Wl2 + bl2     (WMMA bf16, M=64)
  wmma_gemm_bf16<<<dim3(512 / 64, GB / 64), 128, 0, stream>>>(
      zacc, Wl1, bl1, tmlp, GB, 512, 2 * CDIM, 1);
  wmma_gemm_bf16<<<dim3(256 / 64, GB / 64), 128, 0, stream>>>(
      tmlp, Wl2, bl2, (float*)d_out, GB, 256, 512, 0);
}
